// MSAPairWeightedAveraging_9337258901840
// MI455X (gfx1250) — compile-verified, matrix operand order checked by reading, not run on hardware
//
#include <hip/hip_runtime.h>
#include <math.h>

// ---------------------------------------------------------------------------
// Shapes (hardcoded from reference): b=1, s=256, n=768, dm=64, dp=128,
// heads=8, dim_head=32, di=256.
// ---------------------------------------------------------------------------
#define S_DIM   256
#define N_DIM   768
#define DM      64
#define DP      128
#define HEADS   8
#define DHEAD   32
#define DI      256           // HEADS*DHEAD
#define NCOLS   8192          // S_DIM * DHEAD  (columns of the big GEMM)
#define KPANELS 24            // N_DIM / 32 (K panels of the values tensor)

typedef __attribute__((ext_vector_type(16))) __bf16        v16bf;
typedef __attribute__((ext_vector_type(8)))  float         v8f;
typedef __attribute__((ext_vector_type(8)))  unsigned int  v8u;

typedef __attribute__((address_space(3))) unsigned short lds_us_t;

union FragU { v8u u; v16bf b; };

__device__ __forceinline__ unsigned short f2bf(float f) {
  unsigned int u = __float_as_uint(f);
  unsigned int r = u + 0x7fffu + ((u >> 16) & 1u);   // round-to-nearest-even
  return (unsigned short)(r >> 16);
}

__device__ __forceinline__ float wave_sum(float v) {
#pragma unroll
  for (int o = 16; o > 0; o >>= 1) v += __shfl_xor(v, o, 32);
  return v;
}

// Async DMA: global -> LDS, 16 bytes per lane, tracked by ASYNCcnt.
__device__ __forceinline__ void async_b128(lds_us_t* lds, const unsigned short* gaddr) {
  asm volatile("global_load_async_to_lds_b128 %0, %1, off"
               :: "v"(lds), "v"(gaddr) : "memory");
}
__device__ __forceinline__ void wait_async0() {
  asm volatile("s_wait_asynccnt 0" ::: "memory");
}

// ---------------------------------------------------------------------------
// Kernel A: LN(msa) + x @ W_vg[:, 0:256] -> values, bf16, K-panelized layout
//   V[h][j>>5][s*32+d][j&31]   (i.e. [N][K]-major inside each 32-wide K panel)
// so the GEMM's B tiles are async-copied with no transpose.
// One wave per token (s,j). 196608 tokens / 8 waves = 24576 blocks.
// ---------------------------------------------------------------------------
__global__ __launch_bounds__(256) void pwa_ln_values_kernel(
    const float* __restrict__ msa, const float* __restrict__ ln_g,
    const float* __restrict__ ln_b, const float* __restrict__ Wvg,
    unsigned short* __restrict__ Vout)
{
  __shared__ float xs[8][DM];
  const int tid = threadIdx.x, lane = tid & 31, w = tid >> 5;
  const int tok = blockIdx.x * 8 + w;            // s*768 + j
  const float* row = msa + (size_t)tok * DM;

  float x0 = row[lane], x1 = row[lane + 32];
  float mu = wave_sum(x0 + x1) * (1.0f / 64.0f);
  float d0 = x0 - mu, d1 = x1 - mu;
  float var = wave_sum(d0 * d0 + d1 * d1) * (1.0f / 64.0f);
  float inv = rsqrtf(var + 1e-5f);
  xs[w][lane]      = d0 * inv * ln_g[lane]      + ln_b[lane];
  xs[w][lane + 32] = d1 * inv * ln_g[lane + 32] + ln_b[lane + 32];
  __syncthreads();

  const int s = tok / N_DIM, j = tok % N_DIM;
  const int jp = j >> 5, jk = j & 31;
#pragma unroll
  for (int cc = 0; cc < 8; ++cc) {
    const int c = lane + cc * 32;                // 0..255 -> head = cc, d = lane
    float a = 0.f;
#pragma unroll 8
    for (int k = 0; k < DM; ++k) a += xs[w][k] * Wvg[k * 512 + c];
    const int head = c >> 5, d = c & 31;
    Vout[(((size_t)head * KPANELS + jp) * NCOLS + (s * DHEAD + d)) * 32 + jk] = f2bf(a);
  }
}

// ---------------------------------------------------------------------------
// Kernel B1: LN(pairwise) @ W_b + mask -> bias[h][i][j] (fp32).
// One wave per (i,j). 589824 pairs / 8 = 73728 blocks.
// ---------------------------------------------------------------------------
__global__ __launch_bounds__(256) void pwa_pair_bias_kernel(
    const float* __restrict__ pairr, const float* __restrict__ g,
    const float* __restrict__ bvec, const float* __restrict__ Wb,
    const unsigned char* __restrict__ mask, float* __restrict__ bias)
{
  const int tid = threadIdx.x, lane = tid & 31, w = tid >> 5;
  const int pid = blockIdx.x * 8 + w;            // i*768 + j
  const int i = pid / N_DIM, j = pid % N_DIM;
  const float* row = pairr + (size_t)pid * DP;

  float x[4];
#pragma unroll
  for (int q = 0; q < 4; ++q) x[q] = row[lane + q * 32];
  float mu = wave_sum(x[0] + x[1] + x[2] + x[3]) * (1.0f / 128.0f);
  float v = 0.f;
#pragma unroll
  for (int q = 0; q < 4; ++q) { x[q] -= mu; v += x[q] * x[q]; }
  float inv = rsqrtf(wave_sum(v) * (1.0f / 128.0f) + 1e-5f);
#pragma unroll
  for (int q = 0; q < 4; ++q)
    x[q] = x[q] * inv * g[lane + q * 32] + bvec[lane + q * 32];

  float acc[HEADS];
#pragma unroll
  for (int hh = 0; hh < HEADS; ++hh) {
    float a = 0.f;
#pragma unroll
    for (int q = 0; q < 4; ++q) a += x[q] * Wb[(lane + q * 32) * HEADS + hh];
    acc[hh] = wave_sum(a);
  }
  if (lane < HEADS) {
    float bv = mask[j] ? acc[lane] : -3.4028235e38f;
    bias[(size_t)lane * (N_DIM * N_DIM) + (size_t)i * N_DIM + j] = bv;
  }
}

// ---------------------------------------------------------------------------
// Kernel B2: row softmax over j, output bf16 weights W[h][i][j].
// One block per (h,i) row: 6144 blocks x 256 threads (3 j per thread).
// ---------------------------------------------------------------------------
__global__ __launch_bounds__(256) void pwa_softmax_kernel(
    const float* __restrict__ bias, unsigned short* __restrict__ wout)
{
  const int row = blockIdx.x;                    // h*768 + i
  const float* src = bias + (size_t)row * N_DIM;
  unsigned short* dst = wout + (size_t)row * N_DIM;
  const int tid = threadIdx.x;
  __shared__ float red[256];

  float x[3];
#pragma unroll
  for (int q = 0; q < 3; ++q) x[q] = src[tid + q * 256];
  float m = fmaxf(fmaxf(x[0], x[1]), x[2]);
  red[tid] = m; __syncthreads();
  for (int o = 128; o > 0; o >>= 1) {
    if (tid < o) red[tid] = fmaxf(red[tid], red[tid + o]);
    __syncthreads();
  }
  m = red[0]; __syncthreads();

  float e[3], sum = 0.f;
#pragma unroll
  for (int q = 0; q < 3; ++q) { e[q] = __expf(x[q] - m); sum += e[q]; }
  red[tid] = sum; __syncthreads();
  for (int o = 128; o > 0; o >>= 1) {
    if (tid < o) red[tid] += red[tid + o];
    __syncthreads();
  }
  const float rs = 1.0f / red[0];
#pragma unroll
  for (int q = 0; q < 3; ++q) dst[tid + q * 256] = f2bf(e[q] * rs);
}

// ---------------------------------------------------------------------------
// Kernel C: the dominant einsum as bf16 WMMA GEMM, per head:
//   mid[h][i][(s,d)] = sum_j W[h][i][j] * V[h][j][(s,d)]
// Tiles: BM=128, BN=256, BK=32. 8 waves (2x4), each wave 64x64 = 4x4 WMMA
// tiles. Tiles are DMA'd with GLOBAL_LOAD_ASYNC_TO_LDS_B128 (ASYNCcnt) into
// double-buffered LDS so the next panel's copy overlaps the current WMMAs,
// with zero VGPR staging. Row stride 40 ushorts = 80B keeps every async B128
// store 16B-aligned and fragment reads spread across banks.
// ---------------------------------------------------------------------------
#define BM 128
#define BN 256
#define BK 32
#define LPAD 8
#define LSTR (BK + LPAD)      // 40 ushorts per LDS row

__global__ __launch_bounds__(256, 1) void pwa_einsum_kernel(
    const unsigned short* __restrict__ Wgt,     // [8][768][768] bf16
    const unsigned short* __restrict__ Vmat,    // [8][24][8192][32] bf16
    float* __restrict__ mid)                    // [(s*768+i)*256 + h*32 + d]
{
  const int h   = blockIdx.z;
  const int m0  = blockIdx.y * BM;
  const int n0  = blockIdx.x * BN;
  const int tid = threadIdx.x;
  const int lane = tid & 31;
  const int wave = tid >> 5;
  const int half = lane >> 4;                   // 0 / 1
  const int idx  = lane & 15;
  const int wm = (wave >> 2) * 64;              // wave M offset (2 waves in M)
  const int wn = (wave & 3) * 64;               // wave N offset (4 waves in N)

  __shared__ unsigned short As[2][BM][LSTR];
  __shared__ unsigned short Bs[2][BN][LSTR];
  lds_us_t* As3 = (lds_us_t*)&As[0][0][0];
  lds_us_t* Bs3 = (lds_us_t*)&Bs[0][0][0];

  const unsigned short* Ag = Wgt  + (size_t)h * N_DIM * N_DIM;
  const unsigned short* Vg = Vmat + (size_t)h * KPANELS * NCOLS * 32;

  // Thread-invariant chunk coordinates for the async copies (16B = 8 ushorts).
  const int ar0 = (tid) >> 2,            asub = (tid & 3) * 8;          // A it=0
  const int ar1 = (tid + 256) >> 2;                                      // A it=1

  v8f acc[4][4];
#pragma unroll
  for (int mi = 0; mi < 4; ++mi)
#pragma unroll
    for (int bi = 0; bi < 4; ++bi)
#pragma unroll
      for (int r = 0; r < 8; ++r) acc[mi][bi][r] = 0.f;

  // Issue async copy of one K panel (kp) into LDS buffer `buf`.
  auto issue = [&](int buf, int kp) {
    const int k0 = kp * BK;
    lds_us_t* Ab = As3 + (size_t)buf * BM * LSTR;
    lds_us_t* Bb = Bs3 + (size_t)buf * BN * LSTR;
    // A tile: 128 rows x 32 ushorts = 512 chunks, 2 per thread.
    async_b128(Ab + ar0 * LSTR + asub, Ag + (size_t)(m0 + ar0) * N_DIM + k0 + asub);
    async_b128(Ab + ar1 * LSTR + asub, Ag + (size_t)(m0 + ar1) * N_DIM + k0 + asub);
    // B tile: 256 rows x 32 ushorts = 1024 chunks, 4 per thread.
    const unsigned short* Bp = Vg + (size_t)kp * NCOLS * 32;
#pragma unroll
    for (int it = 0; it < 4; ++it) {
      const int c = tid + it * 256;
      const int n = c >> 2, sub = (c & 3) * 8;
      async_b128(Bb + n * LSTR + sub, Bp + (size_t)(n0 + n) * 32 + sub);
    }
  };

  issue(0, 0);

  for (int kt = 0; kt < KPANELS; ++kt) {
    const int cur = kt & 1;
    wait_async0();                 // this wave's copies for `cur` are in LDS
    __syncthreads();               // everyone's copies for `cur` are in LDS
    if (kt + 1 < KPANELS) issue(cur ^ 1, kt + 1);   // overlap DMA with WMMA

    // Fragment K-pair index per ISA 16-bit layout:
    // vgpr r<4 : pairs {r + 4*half}, vgpr r>=4 : pairs {8 + (r-4) + 4*half}
    FragU bfr[4];
#pragma unroll
    for (int bi = 0; bi < 4; ++bi) {
      const unsigned int* brow =
          reinterpret_cast<const unsigned int*>(&Bs[cur][wn + bi * 16 + idx][0]);
#pragma unroll
      for (int r = 0; r < 8; ++r)
        bfr[bi].u[r] = brow[(r & 3) + (half << 2) + ((r >> 2) << 3)];
    }
#pragma unroll
    for (int mi = 0; mi < 4; ++mi) {
      FragU afr;
      const unsigned int* arow =
          reinterpret_cast<const unsigned int*>(&As[cur][wm + mi * 16 + idx][0]);
#pragma unroll
      for (int r = 0; r < 8; ++r)
        afr.u[r] = arow[(r & 3) + (half << 2) + ((r >> 2) << 3)];
#pragma unroll
      for (int bi = 0; bi < 4; ++bi) {
        acc[mi][bi] = __builtin_amdgcn_wmma_f32_16x16x32_bf16(
            false, afr.b, false, bfr[bi].b, (short)0, acc[mi][bi], false, false);
      }
    }
    __syncthreads();               // all waves done reading `cur` before reuse
  }

  // Epilogue: C/D layout — lane: n = idx, row m = r + 8*half.
#pragma unroll
  for (int mi = 0; mi < 4; ++mi) {
#pragma unroll
    for (int bi = 0; bi < 4; ++bi) {
      const int ncol = n0 + wn + bi * 16 + idx;   // s*32 + d
      const int s = ncol >> 5, d = ncol & 31;
      float* dst = mid + (size_t)s * N_DIM * DI + (size_t)(h * DHEAD + d);
#pragma unroll
      for (int r = 0; r < 8; ++r) {
        const int i = m0 + wm + mi * 16 + r + 8 * half;
        dst[(size_t)i * DI] = acc[mi][bi][r];
      }
    }
  }
}

// ---------------------------------------------------------------------------
// Kernel D: epilogue. Recompute gates (LN + W_vg[:,256:512] + sigmoid),
// multiply with mid row, project through W_out. One wave per row (s,i).
// ---------------------------------------------------------------------------
__global__ __launch_bounds__(256) void pwa_epilogue_kernel(
    const float* __restrict__ msa, const float* __restrict__ ln_g,
    const float* __restrict__ ln_b, const float* __restrict__ Wvg,
    const float* __restrict__ mid, const float* __restrict__ Wout,
    float* __restrict__ out)
{
  __shared__ float xs[8][DM];
  __shared__ float av[8][DI];
  const int tid = threadIdx.x, lane = tid & 31, w = tid >> 5;
  const int tok = blockIdx.x * 8 + w;            // s*768 + i
  const float* row = msa + (size_t)tok * DM;

  float x0 = row[lane], x1 = row[lane + 32];
  float mu = wave_sum(x0 + x1) * (1.0f / 64.0f);
  float d0 = x0 - mu, d1 = x1 - mu;
  float var = wave_sum(d0 * d0 + d1 * d1) * (1.0f / 64.0f);
  float inv = rsqrtf(var + 1e-5f);
  xs[w][lane]      = d0 * inv * ln_g[lane]      + ln_b[lane];
  xs[w][lane + 32] = d1 * inv * ln_g[lane + 32] + ln_b[lane + 32];
  __syncthreads();

  const float* midrow = mid + (size_t)tok * DI;
#pragma unroll
  for (int cc = 0; cc < 8; ++cc) {
    const int c = lane + cc * 32;                // 0..255 = h*32+d
    float aG = 0.f;
#pragma unroll 8
    for (int k = 0; k < DM; ++k) aG += xs[w][k] * Wvg[k * 512 + 256 + c];
    const float gate = 1.0f / (1.0f + __expf(-aG));
    av[w][c] = midrow[c] * gate;
  }
  __syncthreads();

  float o0 = 0.f, o1 = 0.f;
  for (int k = 0; k < DI; ++k) {
    const float a = av[w][k];
    o0 += a * Wout[k * DM + lane];
    o1 += a * Wout[k * DM + lane + 32];
  }
  out[(size_t)tok * DM + lane]      = o0;
  out[(size_t)tok * DM + lane + 32] = o1;
}

// ---------------------------------------------------------------------------
// Workspace layout (bytes):
//   values bf16  [8][24][8192][32] : 100,663,296
//   bias   fp32  [8][768][768]     :  18,874,368
//   wgt    bf16  [8][768][768]     :   9,437,184
//   mid    fp32  [256*768][256]    : 201,326,592
// Total ~330.3 MB.
// ---------------------------------------------------------------------------
extern "C" void kernel_launch(void* const* d_in, const int* in_sizes, int n_in,
                              void* d_out, int out_size, void* d_ws, size_t ws_size,
                              hipStream_t stream) {
  (void)in_sizes; (void)n_in; (void)out_size; (void)ws_size;
  const float*         msa      = (const float*)d_in[0];
  const float*         pairr    = (const float*)d_in[1];
  const unsigned char* mask     = (const unsigned char*)d_in[2];
  const float*         ln_msa_g = (const float*)d_in[3];
  const float*         ln_msa_b = (const float*)d_in[4];
  const float*         Wvg      = (const float*)d_in[5];
  const float*         ln_p_g   = (const float*)d_in[6];
  const float*         ln_p_b   = (const float*)d_in[7];
  const float*         Wb       = (const float*)d_in[8];
  const float*         Wout     = (const float*)d_in[9];
  float* out = (float*)d_out;

  char* ws = (char*)d_ws;
  unsigned short* vals = (unsigned short*)(ws + 0);
  float*          bias = (float*)(ws + 100663296ull);
  unsigned short* wgt  = (unsigned short*)(ws + 119537664ull);
  float*          mid  = (float*)(ws + 128974848ull);

  pwa_ln_values_kernel<<<24576, 256, 0, stream>>>(msa, ln_msa_g, ln_msa_b, Wvg, vals);
  pwa_pair_bias_kernel<<<73728, 256, 0, stream>>>(pairr, ln_p_g, ln_p_b, Wb, mask, bias);
  pwa_softmax_kernel<<<6144, 256, 0, stream>>>(bias, wgt);
  pwa_einsum_kernel<<<dim3(NCOLS / BN, N_DIM / BM, HEADS), 256, 0, stream>>>(wgt, vals, mid);
  pwa_epilogue_kernel<<<24576, 256, 0, stream>>>(msa, ln_msa_g, ln_msa_b, Wvg, mid, Wout, out);
}
